// LSTMCell_30640296690430
// MI455X (gfx1250) — matrix operand order repaired and checked by reading, not compiled, not run on hardware
//
#include <hip/hip_runtime.h>
#include <hip/hip_bf16.h>

// ---------------------------------------------------------------------------
// Fused LSTM cell forward for MI455X (gfx1250, wave32, WMMA).
//
// z = [x | h_tm1] @ [W ; U] + b   (A: [B,256] f32 -> f16, W: [256,512] f32 -> f16)
// gates -> tanh -> c = f*c_tm1 + i*g ; h = o*tanh(c)
//
// Roofline: ~320 MB HBM @ 23.3 TB/s ≈ 13.7 us target; 34.4 GFLOP needs
// ~2.5 PFLOPS -> f16 WMMA (16x16x32, f32 acc). Each wave processes TWO
// 16-row tiles against every B fragment (ds:wmma = 1:1, LDS off the
// critical path). B fragments are software-pipelined exactly ONE K-step
// ahead, with sched_barriers bounding the scheduler so the wave stays
// near the 256-VGPR fast window (no s_set_vgpr_msb churn).
// ---------------------------------------------------------------------------

typedef __attribute__((ext_vector_type(16))) _Float16 v16h;
typedef __attribute__((ext_vector_type(8)))  _Float16 v8h;
typedef __attribute__((ext_vector_type(8)))  float    v8f;

#define UNITS      128
#define INPUT_DIM  128
#define KDIM       256      // INPUT_DIM + UNITS (fused GEMM K)
#define NDIM       512      // 4 * UNITS
#define M_BLK      256      // batch rows per workgroup
#define THREADS    256      // 8 waves (wave32)
#define WSTRIDE    264      // KDIM + 8 halves pad: rows 528B (16B aligned),
                            // consecutive n -> +2 banks -> <=2-way conflicts

__device__ __forceinline__ float fast_tanh(float x) {
#if __has_builtin(__builtin_amdgcn_tanhf)
  return __builtin_amdgcn_tanhf(x);       // v_tanh_f32 (gfx1250 TRANS op)
#else
  return tanhf(x);
#endif
}

__device__ __forceinline__ void sched_fence() {
#if __has_builtin(__builtin_amdgcn_sched_barrier)
  __builtin_amdgcn_sched_barrier(0);      // bound scheduler lookahead
#else
  asm volatile("" ::: "memory");
#endif
}

// B fragment (32x16 f16, wave32): lane -> column n = lane&15,
// elements 0..7 -> K = kb+hi*8..+7, elements 8..15 -> K = kb+16+hi*8..+7
// (hi*8 folded into the base pointer). 2x ds_load_b128.
__device__ __forceinline__ v16h load_bfrag(const _Float16* base, int s) {
  const v8h* p = (const v8h*)(base + s * 32);
  v8h lo = p[0];
  v8h hv = p[2];
  return __builtin_shufflevector(lo, hv, 0, 1, 2, 3, 4, 5, 6, 7,
                                         8, 9, 10, 11, 12, 13, 14, 15);
}

// A fragments for one 16-row tile: 16x256 of [x|h] as 8 x v16h.
// A layout (16-bit, 16x32): lane row m = lane&15; kb = hi*8;
// elements 0..7 -> K = 32s+kb..+7, elements 8..15 -> K = 32s+kb+16..+23.
__device__ __forceinline__ void load_afrags(v16h* afr, const float* __restrict__ x,
                                            const float* __restrict__ h_tm1,
                                            int rowA, int hi) {
#pragma unroll
  for (int s = 0; s < 8; ++s) {
    const float* src = (s < 4) ? x : h_tm1;
    const int k0 = (s & 3) * 32 + hi * 8;
    const float4* p = (const float4*)(src + (size_t)rowA * INPUT_DIM + k0);
    const float4 f0 = p[0], f1 = p[1], f2 = p[4], f3 = p[5];
    v16h a;
    a[0]  = (_Float16)f0.x; a[1]  = (_Float16)f0.y;
    a[2]  = (_Float16)f0.z; a[3]  = (_Float16)f0.w;
    a[4]  = (_Float16)f1.x; a[5]  = (_Float16)f1.y;
    a[6]  = (_Float16)f1.z; a[7]  = (_Float16)f1.w;
    a[8]  = (_Float16)f2.x; a[9]  = (_Float16)f2.y;
    a[10] = (_Float16)f2.z; a[11] = (_Float16)f2.w;
    a[12] = (_Float16)f3.x; a[13] = (_Float16)f3.y;
    a[14] = (_Float16)f3.z; a[15] = (_Float16)f3.w;
    afr[s] = a;
  }
}

__global__ __launch_bounds__(THREADS, 2)
void lstm_fused_kernel(const float* __restrict__ x,
                       const float* __restrict__ h_tm1,
                       const float* __restrict__ c_tm1,
                       const float* __restrict__ Wk,   // [128,512] row-major
                       const float* __restrict__ Wr,   // [128,512] row-major
                       const float* __restrict__ bias, // [512]
                       float* __restrict__ out_h,      // [B,128]
                       float* __restrict__ out_c) {    // [B,128]
  extern __shared__ char smem[];
  _Float16* wlds = (_Float16*)smem;                                   // 264 KB
  float*    blds = (float*)(smem + (size_t)NDIM * WSTRIDE * sizeof(_Float16));

  const int tid = threadIdx.x;

  // ---- stage bias (f32) ----
  blds[tid]           = bias[tid];
  blds[tid + THREADS] = bias[tid + THREADS];

  // ---- stage fused weights f32 -> f16, column-major wlds[n*WSTRIDE + k] ----
  // Lanes take consecutive n (coalesced global b32 reads per j), one 16B LDS
  // store per thread (<=2-way bank conflict).
#pragma unroll 1
  for (int it = 0; it < (KDIM / 8) * NDIM / THREADS; ++it) {          // 64 iters
    const int pair = it * THREADS + tid;          // 0..16383
    const int n    = pair & (NDIM - 1);           // fast index: coalesced
    const int kb   = (pair >> 9) << 3;            // K block of 8 (uniform/iter)
    v8h pk;
#pragma unroll
    for (int j = 0; j < 8; ++j) {
      const int kk = kb + j;
      const float w = (kk < INPUT_DIM) ? Wk[kk * NDIM + n]
                                       : Wr[(kk - INPUT_DIM) * NDIM + n];
      pk[j] = (_Float16)w;
    }
    *(v8h*)(wlds + (size_t)n * WSTRIDE + kb) = pk;
  }
  __syncthreads();

  const int lane   = tid & 31;
  const int wave   = tid >> 5;
  const int lane16 = lane & 15;
  const int hi     = lane >> 4;                   // half-wave id (0/1)

  // Each wave owns two 16-row tiles, processed JOINTLY so every B fragment
  // read from LDS feeds two WMMAs.
  const int row0 = blockIdx.x * M_BLK + wave * 32;   // tile 0
  const int row1 = row0 + 16;                        // tile 1

  // Load A fragments in two fenced phases (caps live f32 temps so staging
  // does not set the register high-water mark).
  v16h af0[8], af1[8];
  load_afrags(af0, x, h_tm1, row0 + lane16, hi);
  sched_fence();
  load_afrags(af1, x, h_tm1, row1 + lane16, hi);
  sched_fence();

#pragma unroll 1
  for (int ct = 0; ct < 8; ++ct) {
    const int ncol = ct * 16 + lane16;            // column within UNITS
    // Per-gate base pointers; ds_loads use immediate offsets s*64(+32) B.
    const _Float16* bg0 = wlds + (0 * UNITS + ncol) * WSTRIDE + hi * 8;
    const _Float16* bg1 = wlds + (1 * UNITS + ncol) * WSTRIDE + hi * 8;
    const _Float16* bg2 = wlds + (2 * UNITS + ncol) * WSTRIDE + hi * 8;
    const _Float16* bg3 = wlds + (3 * UNITS + ncol) * WSTRIDE + hi * 8;

    // accumulators: [tile][gate]  (gates i, f, g, o)
    v8f a00 = {}, a01 = {}, a02 = {}, a03 = {};
    v8f a10 = {}, a11 = {}, a12 = {}, a13 = {};

    // One-K-step-ahead pipeline: prefetch loads interleaved between the
    // WMMA pairs; sched_fence at iteration end stops deeper hoisting.
    v16h b0 = load_bfrag(bg0, 0);
    v16h b1 = load_bfrag(bg1, 0);
    v16h b2 = load_bfrag(bg2, 0);
    v16h b3 = load_bfrag(bg3, 0);
    sched_fence();

#pragma unroll
    for (int s = 0; s < 8; ++s) {                 // K steps of 32
      v16h n0, n1, n2, n3;
      if (s < 7) n0 = load_bfrag(bg0, s + 1);
      a00 = __builtin_amdgcn_wmma_f32_16x16x32_f16(false, af0[s], false, b0,
                                                   (short)0, a00, false, false);
      a10 = __builtin_amdgcn_wmma_f32_16x16x32_f16(false, af1[s], false, b0,
                                                   (short)0, a10, false, false);
      if (s < 7) n1 = load_bfrag(bg1, s + 1);
      a01 = __builtin_amdgcn_wmma_f32_16x16x32_f16(false, af0[s], false, b1,
                                                   (short)0, a01, false, false);
      a11 = __builtin_amdgcn_wmma_f32_16x16x32_f16(false, af1[s], false, b1,
                                                   (short)0, a11, false, false);
      if (s < 7) n2 = load_bfrag(bg2, s + 1);
      a02 = __builtin_amdgcn_wmma_f32_16x16x32_f16(false, af0[s], false, b2,
                                                   (short)0, a02, false, false);
      a12 = __builtin_amdgcn_wmma_f32_16x16x32_f16(false, af1[s], false, b2,
                                                   (short)0, a12, false, false);
      if (s < 7) n3 = load_bfrag(bg3, s + 1);
      a03 = __builtin_amdgcn_wmma_f32_16x16x32_f16(false, af0[s], false, b3,
                                                   (short)0, a03, false, false);
      a13 = __builtin_amdgcn_wmma_f32_16x16x32_f16(false, af1[s], false, b3,
                                                   (short)0, a13, false, false);
      if (s < 7) { b0 = n0; b1 = n1; b2 = n2; b3 = n3; }
      sched_fence();                              // cap lookahead to 1 K-step
    }

    // ---- epilogue: full LSTM gating in registers ----
    // C/D layout: element r <-> row (r + 8*hi), column lane&15.
    const float bi = blds[0 * UNITS + ncol];
    const float bf = blds[1 * UNITS + ncol];
    const float bc = blds[2 * UNITS + ncol];
    const float bo = blds[3 * UNITS + ncol];

#pragma unroll
    for (int tile = 0; tile < 2; ++tile) {
      const v8f& gi = tile ? a10 : a00;
      const v8f& gf = tile ? a11 : a01;
      const v8f& gc = tile ? a12 : a02;
      const v8f& go = tile ? a13 : a03;
      const int rowE = (tile ? row1 : row0) + hi * 8;
#pragma unroll
      for (int r = 0; r < 8; ++r) {
        const int idx = (rowE + r) * UNITS + ncol;   // < 2^24: 32-bit math
        const float c_old = c_tm1[idx];
        const float ig = fast_tanh(gi[r] + bi);   // recurrent_activation = tanh
        const float fg = fast_tanh(gf[r] + bf);
        const float gg = fast_tanh(gc[r] + bc);
        const float og = fast_tanh(go[r] + bo);
        const float cn = fg * c_old + ig * gg;
        const float hn = og * fast_tanh(cn);
        out_h[idx] = hn;
        out_c[idx] = cn;
      }
    }
  }
}

extern "C" void kernel_launch(void* const* d_in, const int* in_sizes, int n_in,
                              void* d_out, int out_size, void* d_ws, size_t ws_size,
                              hipStream_t stream) {
  (void)in_sizes; (void)n_in; (void)d_ws; (void)ws_size; (void)out_size;
  const float* x    = (const float*)d_in[0];
  const float* h    = (const float*)d_in[1];
  const float* c    = (const float*)d_in[2];
  const float* Wk   = (const float*)d_in[3];
  const float* Wr   = (const float*)d_in[4];
  const float* bias = (const float*)d_in[5];

  const int BATCH = 131072;
  float* out_h = (float*)d_out;
  float* out_c = out_h + (size_t)BATCH * UNITS;

  const size_t shmem = (size_t)NDIM * WSTRIDE * sizeof(_Float16)   // 270336 B
                     + (size_t)NDIM * sizeof(float);               // + 2048 B
  // Allow >default dynamic LDS (WGP supports 320 KB per workgroup).
  (void)hipFuncSetAttribute((const void*)lstm_fused_kernel,
                            hipFuncAttributeMaxDynamicSharedMemorySize,
                            (int)shmem);

  dim3 grid(BATCH / M_BLK);   // 512 workgroups
  dim3 block(THREADS);        // 8 waves
  lstm_fused_kernel<<<grid, block, shmem, stream>>>(x, h, c, Wk, Wr, bias,
                                                    out_h, out_c);
}
